// MambaBlock_21663815041677
// MI455X (gfx1250) — matrix operand order repaired and checked by reading, not compiled, not run on hardware
//
#include <hip/hip_runtime.h>
#include <hip/hip_bf16.h>
#include <math.h>

#define D_MODEL   768
#define D_STATE   128
#define HEADDIM   48
#define D_INNER   1536
#define NHEADS    32
#define CONV_DIM  1792
#define D_IN_PROJ 3360
#define BATCH     2
#define SEQ       1024
#define ROWS      (BATCH * SEQ)   // 2048
#define Q         64              // scan chunk length
#define NC        (SEQ / Q)       // 16 chunks

typedef __attribute__((ext_vector_type(16))) __bf16 v16bf;
typedef __attribute__((ext_vector_type(8)))  float  v8f;

union FragBF { v16bf v; uint4 q[2]; };

#define WMMA_BF16(a, b, c) \
  __builtin_amdgcn_wmma_f32_16x16x32_bf16(false, (a), false, (b), (short)0, (c), false, false)

__device__ __forceinline__ unsigned short f2bf(float f) {
  unsigned int u = __float_as_uint(f);
  unsigned int r = u + 0x7FFFu + ((u >> 16) & 1u);   // round-to-nearest-even
  return (unsigned short)(r >> 16);
}

// ---------------------------------------------------------------- fp32 -> bf16
__global__ void k_f32_to_bf16(const float* __restrict__ in,
                              unsigned short* __restrict__ out, int n) {
  int i = blockIdx.x * blockDim.x + threadIdx.x;
  if (i < n) out[i] = f2bf(in[i]);
}

// ---------------------------------------------------------------- bf16 WMMA GEMM
// Out[M,N] = A[M,K] * W[N,K]^T (+ resid). Block 128 thr (4 waves), tile 64x32.
__global__ __launch_bounds__(128) void k_gemm_bf16(
    const unsigned short* __restrict__ A, const unsigned short* __restrict__ W,
    const float* __restrict__ resid, float* __restrict__ Out,
    int M, int N, int K) {
  const int lane = threadIdx.x & 31;
  const int wave = threadIdx.x >> 5;
  const int m0 = blockIdx.y * 64 + wave * 16;
  const int n0 = blockIdx.x * 32;
  const int lm = lane & 15;
  const int hi = (lane >> 4) & 1;

  v8f acc0 = {}; v8f acc1 = {};
  const unsigned short* arow  = A + (size_t)(m0 + lm) * K + hi * 16;
  const unsigned short* brow0 = W + (size_t)(n0 + lm) * K + hi * 16;
  const unsigned short* brow1 = W + (size_t)(n0 + 16 + lm) * K + hi * 16;

  for (int k = 0; k < K; k += 32) {
    FragBF a, b0, b1;
    a.q[0]  = *(const uint4*)(arow  + k);
    a.q[1]  = *(const uint4*)(arow  + k + 8);
    b0.q[0] = *(const uint4*)(brow0 + k);
    b0.q[1] = *(const uint4*)(brow0 + k + 8);
    b1.q[0] = *(const uint4*)(brow1 + k);
    b1.q[1] = *(const uint4*)(brow1 + k + 8);
    acc0 = WMMA_BF16(a.v, b0.v, acc0);
    acc1 = WMMA_BF16(a.v, b1.v, acc1);
  }
  #pragma unroll
  for (int i = 0; i < 8; ++i) {
    int row = m0 + i + hi * 8;
    size_t o0 = (size_t)row * N + n0 + lm;
    size_t o1 = o0 + 16;
    float r0 = acc0[i], r1 = acc1[i];
    if (resid) { r0 += resid[o0]; r1 += resid[o1]; }
    Out[o0] = r0; Out[o1] = r1;
  }
}

// ---------------------------------------------------------------- depthwise conv4 + SiLU
__global__ void k_conv_silu(const float* __restrict__ zx,
                            const float* __restrict__ cw,
                            const float* __restrict__ cb,
                            float* __restrict__ out) {
  int idx = blockIdx.x * blockDim.x + threadIdx.x;
  if (idx >= ROWS * CONV_DIM) return;
  int c  = idx % CONV_DIM;
  int bl = idx / CONV_DIM;
  int l = bl % SEQ, b = bl / SEQ;
  float acc = cb[c];
  #pragma unroll
  for (int k = 0; k < 4; ++k) {
    int lk = l + k - 3;
    if (lk >= 0)
      acc += zx[((size_t)(b * SEQ + lk)) * D_IN_PROJ + D_INNER + c] * cw[c * 4 + k];
  }
  out[(size_t)bl * CONV_DIM + c] = acc / (1.0f + __expf(-acc));
}

// ---------------------------------------------------------------- dt = softplus(raw + bias)
__global__ void k_dt(const float* __restrict__ zx, const float* __restrict__ dt_bias,
                     float* __restrict__ dtb) {
  int idx = blockIdx.x * blockDim.x + threadIdx.x;
  if (idx >= ROWS * NHEADS) return;
  int h = idx % NHEADS, bl = idx / NHEADS;
  float raw = zx[(size_t)bl * D_IN_PROJ + (D_IN_PROJ - NHEADS) + h] + dt_bias[h];
  dtb[idx] = (raw > 20.f) ? raw : log1pf(__expf(raw));
}

// ---------------------------------------------------------------- chunked scan, intra-chunk
// One block per (b,h,chunk); 256 threads (8 waves). All matrix work via WMMA:
//   G = C B^T (64x64,K=128) -> M = mask(G)*exp(s[l]-s[i]) (bf16, LDS)
//   Y_intra = M Xdt (64x48,K=64);  S_c = XdtT (w*B)^T (48x128,K=64)
__global__ __launch_bounds__(256) void k_chunk_intra(
    const float* __restrict__ conv,            // f32 conv out (B,L,1792)
    const unsigned short* __restrict__ convb,  // bf16 conv out
    const float* __restrict__ dtb,             // (B,L,32)
    const float* __restrict__ A_log,
    const float* __restrict__ Dvec,
    unsigned short* __restrict__ csbuf,        // (b,h,c,Q,128) bf16: C*exp(s[l])
    float* __restrict__ scbuf,                 // (b,h,c,48,128) f32 chunk state
    float* __restrict__ decayc,                // (b,h,c) exp(s[Q-1])
    float* __restrict__ ybuf)                  // (B,L,1536)
{
  __shared__ __align__(16) unsigned short sBt[D_STATE][Q + 8];   // (n, i) = w_i*B_i[n]
  __shared__ __align__(16) unsigned short sXdtT[HEADDIM][Q + 8]; // (p, i) = xdt_i[p]
  __shared__ __align__(16) unsigned short sM[Q][Q + 8];          // (l, i)
  __shared__ float s_s[Q];
  __shared__ float s_w[Q];

  const int bhc = blockIdx.x;
  const int c = bhc & (NC - 1);
  const int h = (bhc >> 4) & (NHEADS - 1);
  const int b = bhc >> 9;
  const int t = threadIdx.x;
  const int lane = t & 31, wave = t >> 5;
  const int lm = lane & 15, hi = (lane >> 4) & 1;
  const size_t rowbase = (size_t)(b * SEQ + c * Q);
  const float Dh = Dvec[h];

  // ---- Phase A: log-decay prefix sums (A < 0 => s monotone non-increasing, exps <= 1)
  if (t == 0) {
    float A = -__expf(A_log[h]);
    float s = 0.f;
    for (int i = 0; i < Q; ++i) {
      s += dtb[(rowbase + i) * NHEADS + h] * A;
      s_s[i] = s;
    }
  }
  __syncthreads();
  if (t < Q) s_w[t] = __expf(s_s[Q - 1] - s_s[t]);
  if (t == 0) decayc[bhc] = __expf(s_s[Q - 1]);
  __syncthreads();

  // ---- Phase A2: build LDS operands + scaled-C global store
  {
    int l = t >> 2;
    int n0 = (t & 3) * 32;
    const size_t rowc = (rowbase + l) * CONV_DIM;
    float wl = s_w[l];
    float el = __expf(s_s[l]);
    for (int i = 0; i < 32; ++i) {
      int n = n0 + i;
      float bv = conv[rowc + D_INNER + n];
      sBt[n][l] = f2bf(bv * wl);
      float cv = conv[rowc + D_INNER + D_STATE + n];
      csbuf[((size_t)bhc * Q + l) * D_STATE + n] = f2bf(cv * el);
    }
  }
  if (t < 192) {
    int l = t / 3;
    int p0 = (t % 3) * 16;
    const size_t rowc = (rowbase + l) * CONV_DIM;
    float dt = dtb[(rowbase + l) * NHEADS + h];
    for (int i = 0; i < 16; ++i) {
      int p = p0 + i;
      sXdtT[p][l] = f2bf(conv[rowc + h * HEADDIM + p] * dt);
    }
  }
  __syncthreads();

  // ---- Phase B: G = C B^T, then masked decay -> sM (bf16)
  for (int tile = wave; tile < 16; tile += 8) {
    int m0 = (tile >> 2) * 16;   // l
    int n0 = (tile & 3) * 16;    // i
    v8f acc = {};
    const unsigned short* arow = convb + (rowbase + m0 + lm) * CONV_DIM + (D_INNER + D_STATE) + hi * 16;
    const unsigned short* wrow = convb + (rowbase + n0 + lm) * CONV_DIM + D_INNER + hi * 16;
    for (int k = 0; k < D_STATE; k += 32) {
      FragBF a, w;
      a.q[0] = *(const uint4*)(arow + k);
      a.q[1] = *(const uint4*)(arow + k + 8);
      w.q[0] = *(const uint4*)(wrow + k);
      w.q[1] = *(const uint4*)(wrow + k + 8);
      acc = WMMA_BF16(a.v, w.v, acc);
    }
    #pragma unroll
    for (int i = 0; i < 8; ++i) {
      int l  = m0 + i + hi * 8;
      int ii = n0 + lm;
      float v = (ii <= l) ? acc[i] * __expf(s_s[l] - s_s[ii]) : 0.f;
      sM[l][ii] = f2bf(v);
    }
  }
  __syncthreads();

  // ---- Phase C: Y_intra = M @ Xdt   (+ D*xs, written to ybuf)
  for (int tile = wave; tile < 12; tile += 8) {
    int m0 = (tile / 3) * 16;    // l
    int n0 = (tile % 3) * 16;    // p
    v8f acc = {};
    const unsigned short* arow = &sM[m0 + lm][hi * 16];
    const unsigned short* wrow = &sXdtT[n0 + lm][hi * 16];
    for (int k = 0; k < Q; k += 32) {
      FragBF a, w;
      a.q[0] = *(const uint4*)(arow + k);
      a.q[1] = *(const uint4*)(arow + k + 8);
      w.q[0] = *(const uint4*)(wrow + k);
      w.q[1] = *(const uint4*)(wrow + k + 8);
      acc = WMMA_BF16(a.v, w.v, acc);
    }
    #pragma unroll
    for (int i = 0; i < 8; ++i) {
      int l = m0 + i + hi * 8;
      int p = n0 + lm;
      float xs = conv[(rowbase + l) * CONV_DIM + h * HEADDIM + p];
      ybuf[(rowbase + l) * D_INNER + h * HEADDIM + p] = acc[i] + Dh * xs;
    }
  }

  // ---- Phase D: S_c = XdtT @ (w*B)^T  (48x128, K=64)
  for (int tile = wave; tile < 24; tile += 8) {
    int m0 = (tile / 8) * 16;    // p
    int n0 = (tile % 8) * 16;    // n
    v8f acc = {};
    const unsigned short* arow = &sXdtT[m0 + lm][hi * 16];
    const unsigned short* wrow = &sBt[n0 + lm][hi * 16];
    for (int k = 0; k < Q; k += 32) {
      FragBF a, w;
      a.q[0] = *(const uint4*)(arow + k);
      a.q[1] = *(const uint4*)(arow + k + 8);
      w.q[0] = *(const uint4*)(wrow + k);
      w.q[1] = *(const uint4*)(wrow + k + 8);
      acc = WMMA_BF16(a.v, w.v, acc);
    }
    #pragma unroll
    for (int i = 0; i < 8; ++i) {
      int p = m0 + i + hi * 8;
      int n = n0 + lm;
      scbuf[(size_t)bhc * (HEADDIM * D_STATE) + p * D_STATE + n] = acc[i];
    }
  }
}

// ---------------------------------------------------------------- chunk recurrence (16 steps)
// hprev(c+1) = exp(s63_c) * hprev(c) + S_c ; store hprev(c) bf16 per chunk.
__global__ void k_chunk_state(const float* __restrict__ scbuf,
                              const float* __restrict__ decayc,
                              unsigned short* __restrict__ hprevb) {
  int idx = blockIdx.x * blockDim.x + threadIdx.x;
  if (idx >= BATCH * NHEADS * HEADDIM * D_STATE) return;
  int pn = idx % (HEADDIM * D_STATE);
  int bh = idx / (HEADDIM * D_STATE);
  float hp = 0.f;
  for (int c = 0; c < NC; ++c) {
    size_t o = ((size_t)bh * NC + c) * (HEADDIM * D_STATE) + pn;
    hprevb[o] = f2bf(hp);
    hp = decayc[bh * NC + c] * hp + scbuf[o];
  }
}

// ---------------------------------------------------------------- inter-chunk output
// Y_inter = (C*exp(s[l])) @ hprev^T  (64x48, K=128), accumulated into ybuf.
__global__ __launch_bounds__(128) void k_chunk_inter(
    const unsigned short* __restrict__ csbuf,
    const unsigned short* __restrict__ hprevb,
    float* __restrict__ ybuf) {
  const int bhc = blockIdx.x;
  const int c = bhc & (NC - 1);
  const int h = (bhc >> 4) & (NHEADS - 1);
  const int b = bhc >> 9;
  const int t = threadIdx.x, lane = t & 31, wave = t >> 5;
  const int lm = lane & 15, hi = (lane >> 4) & 1;
  const size_t rowbase = (size_t)(b * SEQ + c * Q);
  const unsigned short* Cs = csbuf + (size_t)bhc * Q * D_STATE;
  const unsigned short* Hp = hprevb + (size_t)bhc * (HEADDIM * D_STATE);

  for (int tile = wave; tile < 12; tile += 4) {
    int m0 = (tile / 3) * 16;    // l
    int n0 = (tile % 3) * 16;    // p
    v8f acc = {};
    const unsigned short* arow = Cs + (size_t)(m0 + lm) * D_STATE + hi * 16;
    const unsigned short* wrow = Hp + (size_t)(n0 + lm) * D_STATE + hi * 16;
    for (int k = 0; k < D_STATE; k += 32) {
      FragBF a, w;
      a.q[0] = *(const uint4*)(arow + k);
      a.q[1] = *(const uint4*)(arow + k + 8);
      w.q[0] = *(const uint4*)(wrow + k);
      w.q[1] = *(const uint4*)(wrow + k + 8);
      acc = WMMA_BF16(a.v, w.v, acc);
    }
    #pragma unroll
    for (int i = 0; i < 8; ++i) {
      int l = m0 + i + hi * 8;
      int p = n0 + lm;
      size_t o = (rowbase + l) * D_INNER + h * HEADDIM + p;
      ybuf[o] += acc[i];
    }
  }
}

// ---------------------------------------------------------------- yg = rmsnorm(y * silu(z)) -> bf16
__global__ __launch_bounds__(256) void k_gated_rms(
    const float* __restrict__ ybuf, const float* __restrict__ zx,
    const float* __restrict__ rms_w, unsigned short* __restrict__ yg) {
  __shared__ float red[256];
  const int row = blockIdx.x, t = threadIdx.x;
  float v[6]; float ss = 0.f;
  #pragma unroll
  for (int i = 0; i < 6; ++i) {
    int c = t + i * 256;
    float z = zx[(size_t)row * D_IN_PROJ + c];
    float g = z / (1.f + __expf(-z));
    float val = ybuf[(size_t)row * D_INNER + c] * g;
    v[i] = val; ss += val * val;
  }
  red[t] = ss; __syncthreads();
  for (int s = 128; s > 0; s >>= 1) {
    if (t < s) red[t] += red[t + s];
    __syncthreads();
  }
  float scale = rsqrtf(red[0] / (float)D_INNER + 1e-5f);
  #pragma unroll
  for (int i = 0; i < 6; ++i) {
    int c = t + i * 256;
    yg[(size_t)row * D_INNER + c] = f2bf(v[i] * scale * rms_w[c]);
  }
}

// ---------------------------------------------------------------- final LayerNorm over 768
__global__ __launch_bounds__(256) void k_layernorm(
    const float* __restrict__ r, const float* __restrict__ lw,
    const float* __restrict__ lb, float* __restrict__ out) {
  __shared__ float red[256];
  const int row = blockIdx.x, t = threadIdx.x;
  float v[3]; float s = 0.f;
  #pragma unroll
  for (int i = 0; i < 3; ++i) { v[i] = r[(size_t)row * D_MODEL + t + i * 256]; s += v[i]; }
  red[t] = s; __syncthreads();
  for (int st = 128; st > 0; st >>= 1) { if (t < st) red[t] += red[t + st]; __syncthreads(); }
  float mu = red[0] / (float)D_MODEL;
  __syncthreads();
  float sq = 0.f;
  #pragma unroll
  for (int i = 0; i < 3; ++i) { float d = v[i] - mu; sq += d * d; }
  red[t] = sq; __syncthreads();
  for (int st = 128; st > 0; st >>= 1) { if (t < st) red[t] += red[t + st]; __syncthreads(); }
  float inv = rsqrtf(red[0] / (float)D_MODEL + 1e-5f);
  #pragma unroll
  for (int i = 0; i < 3; ++i) {
    int c = t + i * 256;
    out[(size_t)row * D_MODEL + c] = (v[i] - mu) * inv * lw[c] + lb[c];
  }
}

// ================================================================ launch
extern "C" void kernel_launch(void* const* d_in, const int* in_sizes, int n_in,
                              void* d_out, int out_size, void* d_ws, size_t ws_size,
                              hipStream_t stream) {
  const float* x          = (const float*)d_in[0];
  const float* in_proj_w  = (const float*)d_in[1];
  const float* conv_w     = (const float*)d_in[2];
  const float* conv_b     = (const float*)d_in[3];
  const float* dt_bias    = (const float*)d_in[4];
  const float* A_log      = (const float*)d_in[5];
  const float* Dv         = (const float*)d_in[6];
  const float* rms_w      = (const float*)d_in[7];
  const float* out_proj_w = (const float*)d_in[8];
  const float* ln_w       = (const float*)d_in[9];
  const float* ln_b       = (const float*)d_in[10];
  float* out = (float*)d_out;

  char* ws = (char*)d_ws;
  size_t off = 0;
  auto alloc = [&](size_t bytes) -> char* {
    char* p = ws + off;
    off += (bytes + 255) & ~(size_t)255;
    return p;
  };

  unsigned short* x_bf  = (unsigned short*)alloc((size_t)ROWS * D_MODEL * 2);
  unsigned short* wi_bf = (unsigned short*)alloc((size_t)D_IN_PROJ * D_MODEL * 2);
  unsigned short* wo_bf = (unsigned short*)alloc((size_t)D_MODEL * D_INNER * 2);
  float* zx      = (float*)alloc((size_t)ROWS * D_IN_PROJ * 4);
  float* convout = (float*)alloc((size_t)ROWS * CONV_DIM * 4);
  unsigned short* convb = (unsigned short*)alloc((size_t)ROWS * CONV_DIM * 2);
  float* dtb     = (float*)alloc((size_t)ROWS * NHEADS * 4);
  unsigned short* csbuf = (unsigned short*)alloc((size_t)BATCH * NHEADS * NC * Q * D_STATE * 2);
  float* scbuf   = (float*)alloc((size_t)BATCH * NHEADS * NC * HEADDIM * D_STATE * 4);
  float* decayc  = (float*)alloc((size_t)BATCH * NHEADS * NC * 4);
  unsigned short* hprevb = (unsigned short*)alloc((size_t)BATCH * NHEADS * NC * HEADDIM * D_STATE * 2);
  float* ybuf    = (float*)alloc((size_t)ROWS * D_INNER * 4);
  unsigned short* yg_bf = (unsigned short*)alloc((size_t)ROWS * D_INNER * 2);
  float* rbuf    = (float*)alloc((size_t)ROWS * D_MODEL * 4);

  // 1. fp32 -> bf16 converts for GEMM operands
  k_f32_to_bf16<<<(ROWS * D_MODEL + 255) / 256, 256, 0, stream>>>(x, x_bf, ROWS * D_MODEL);
  k_f32_to_bf16<<<(D_IN_PROJ * D_MODEL + 255) / 256, 256, 0, stream>>>(in_proj_w, wi_bf, D_IN_PROJ * D_MODEL);
  k_f32_to_bf16<<<(D_MODEL * D_INNER + 255) / 256, 256, 0, stream>>>(out_proj_w, wo_bf, D_MODEL * D_INNER);

  // 2. in_proj GEMM (2048 x 3360 x 768) via WMMA
  dim3 g1(D_IN_PROJ / 32, ROWS / 64);
  k_gemm_bf16<<<g1, 128, 0, stream>>>(x_bf, wi_bf, nullptr, zx, ROWS, D_IN_PROJ, D_MODEL);

  // 3. conv + SiLU, dt, bf16 copy of conv output
  k_conv_silu<<<(ROWS * CONV_DIM + 255) / 256, 256, 0, stream>>>(zx, conv_w, conv_b, convout);
  k_dt<<<(ROWS * NHEADS + 255) / 256, 256, 0, stream>>>(zx, dt_bias, dtb);
  k_f32_to_bf16<<<(ROWS * CONV_DIM + 255) / 256, 256, 0, stream>>>(convout, convb, ROWS * CONV_DIM);

  // 4. chunk-parallel scan: intra-chunk WMMA work (1024 blocks)
  k_chunk_intra<<<BATCH * NHEADS * NC, 256, 0, stream>>>(
      convout, convb, dtb, A_log, Dv, csbuf, scbuf, decayc, ybuf);

  // 5. 16-step chunk recurrence (parallel over B*H*48*128)
  k_chunk_state<<<(BATCH * NHEADS * HEADDIM * D_STATE + 255) / 256, 256, 0, stream>>>(
      scbuf, decayc, hprevb);

  // 6. inter-chunk contribution via WMMA (1024 blocks)
  k_chunk_inter<<<BATCH * NHEADS * NC, 128, 0, stream>>>(csbuf, hprevb, ybuf);

  // 7. gated RMSNorm -> bf16
  k_gated_rms<<<ROWS, 256, 0, stream>>>(ybuf, zx, rms_w, yg_bf);

  // 8. out_proj GEMM (2048 x 768 x 1536) + residual
  dim3 g2(D_MODEL / 32, ROWS / 64);
  k_gemm_bf16<<<g2, 128, 0, stream>>>(yg_bf, wo_bf, x, rbuf, ROWS, D_MODEL, D_INNER);

  // 9. final LayerNorm -> output
  k_layernorm<<<ROWS, 256, 0, stream>>>(rbuf, ln_w, ln_b, out);
}
